// MultiHeadedAttention_19696720019947
// MI455X (gfx1250) — matrix operand order repaired and checked by reading, not compiled, not run on hardware
//
#include <hip/hip_runtime.h>
#include <hip/hip_bf16.h>

#define BATCH 4
#define SEQ   2048
#define DMODEL 512
#define HEADS 8
#define DK    64
#define NKT   (SEQ / 64)

typedef __attribute__((ext_vector_type(16))) __bf16 v16bf;
typedef __attribute__((ext_vector_type(2)))  __bf16 v2bf;
typedef __attribute__((ext_vector_type(8)))  float  v8f;
typedef __attribute__((ext_vector_type(4)))  unsigned int u32x4;
typedef __attribute__((ext_vector_type(4)))  int i32x4;
typedef __attribute__((ext_vector_type(8)))  int i32x8;

union Frag { v16bf bf; unsigned int u[8]; };

// ---------------- f32 -> bf16 conversion ------------------------------------
__device__ __forceinline__ unsigned int pack_bf16(float a, float b) {
#if __has_builtin(__builtin_amdgcn_cvt_pk_bf16_f32)
    v2bf r = __builtin_amdgcn_cvt_pk_bf16_f32(a, b);
    unsigned int u;
    __builtin_memcpy(&u, &r, 4);
    return u;
#else
    // round-half-up bf16; pack the two high halves with one v_perm_b32
    unsigned int ua = __float_as_uint(a) + 0x8000u;
    unsigned int ub = __float_as_uint(b) + 0x8000u;
    return __builtin_amdgcn_perm(ub, ua, 0x07060302u);
#endif
}
__device__ __forceinline__ unsigned short f2bf_hw(float f) {
#if __has_builtin(__builtin_amdgcn_cvt_pk_bf16_f32)
    return (unsigned short)(pack_bf16(f, 0.0f) & 0xFFFFu);
#else
    return (unsigned short)((__float_as_uint(f) + 0x8000u) >> 16);
#endif
}

__device__ __forceinline__ v8f wmma_bf16(v16bf a, v16bf b, v8f c) {
    return __builtin_amdgcn_wmma_f32_16x16x32_bf16(false, a, false, b, (short)0, c, false, false);
}

// A-matrix fragment (16x32 bf16, M x K) from LDS tile stored [row][k] (k contig).
__device__ __forceinline__ v16bf load_a_frag(const unsigned int* lds, int row_base,
                                             int pitch_dw, int kofs_dw, int lane) {
    int m  = row_base + (lane & 15);
    int hi = lane >> 4;
    const unsigned int* p = lds + m * pitch_dw + kofs_dw;
    Frag f;
#pragma unroll
    for (int v = 0; v < 8; ++v) f.u[v] = p[hi * 8 + v];
    return f.bf;
}

// B-matrix fragment (32x16 bf16, K x N) from LDS tile stored [n][k] (k contig).
__device__ __forceinline__ v16bf load_b_frag(const unsigned int* lds, int col_base,
                                             int pitch_dw, int kofs_dw, int lane) {
    int n  = col_base + (lane & 15);
    int hi = lane >> 4;
    const unsigned int* p = lds + n * pitch_dw + kofs_dw;
    Frag f;
#pragma unroll
    for (int v = 0; v < 8; ++v) f.u[v] = p[hi * 8 + v];
    return f.bf;
}

__device__ __forceinline__ float grp16_max(float v) {
#pragma unroll
    for (int off = 1; off < 16; off <<= 1) v = fmaxf(v, __shfl_xor(v, off));
    return v;
}
__device__ __forceinline__ float grp16_sum(float v) {
#pragma unroll
    for (int off = 1; off < 16; off <<= 1) v += __shfl_xor(v, off);
    return v;
}

// ---------------- Tensor Data Mover: async 2D tile -> LDS --------------------
#if __has_builtin(__builtin_amdgcn_tensor_load_to_lds) && \
    __has_builtin(__builtin_amdgcn_s_wait_tensorcnt)
#define HAVE_TDM 1
#endif

#if defined(HAVE_TDM)
// Async-load a [tile_d1 rows x tile_d0 cols] tile of 2-byte elements starting
// at gaddr (row pitch = stride0 elements) into contiguous LDS at lds_off.
__device__ __forceinline__ void tdm_load_2d(unsigned lds_off, const void* gaddr,
                                            unsigned tensor_d0, unsigned tensor_d1,
                                            unsigned tile_d0, unsigned tile_d1,
                                            unsigned stride0)
{
    unsigned long long ga = (unsigned long long)(uintptr_t)gaddr;
    u32x4 g0 = { 1u,                                   // count=1, user descriptor
                 lds_off,                              // lds_addr
                 (unsigned int)ga,                     // global_addr[31:0]
                 (unsigned int)((ga >> 32) & 0x01FFFFFFull) | (2u << 30) }; // [56:32]|type=2
    i32x8 g1 = { (int)(1u << 16),                                    // data_size=2B
                 (int)(tensor_d0 << 16),                             // tensor_dim0 lo
                 (int)((tensor_d0 >> 16) | (tensor_d1 << 16)),       // dim0 hi | dim1 lo
                 (int)((tensor_d1 >> 16) | (tile_d0 << 16)),         // dim1 hi | tile_dim0
                 (int)(tile_d1 & 0xFFFFu),                           // tile_dim1 (tile_dim2=0)
                 (int)stride0,                                       // dim0_stride[31:0]
                 0, 0 };
    i32x4 z4 = { 0, 0, 0, 0 };
#if defined(__clang_major__) && __clang_major__ >= 23
    i32x8 z8 = { 0, 0, 0, 0, 0, 0, 0, 0 };
    __builtin_amdgcn_tensor_load_to_lds(g0, g1, z4, z4, z8, 0);
#else
    __builtin_amdgcn_tensor_load_to_lds(g0, g1, z4, z4, 0);
#endif
}
#endif

// ---------------------------------------------------------------------------
// Kernel 1: projection GEMM  y = x @ W^T + b, bf16 out in [B,H,S,dk]
// (V transposed to [B,H,dk,S]). grid (M/64, N/64, 3), block 256 (8 waves).
// ---------------------------------------------------------------------------
__global__ __launch_bounds__(256) void proj_kernel(
    const float* __restrict__ key, const float* __restrict__ value,
    const float* __restrict__ query,
    const float* __restrict__ Wk, const float* __restrict__ bk,
    const float* __restrict__ Wv, const float* __restrict__ bv,
    const float* __restrict__ Wq, const float* __restrict__ bq,
    unsigned short* __restrict__ q_up, unsigned short* __restrict__ k_up,
    unsigned short* __restrict__ v_upT)
{
    __shared__ __align__(16) unsigned int XA[64 * 16];   // [m][k] bf16, pitch 16 dw
    __shared__ __align__(16) unsigned int WB[64 * 16];   // [n][k] bf16

    const float* x; const float* W; const float* bias;
    unsigned short* out; bool tr;
    if (blockIdx.z == 0)      { x = query; W = Wq; bias = bq; out = q_up;  tr = false; }
    else if (blockIdx.z == 1) { x = key;   W = Wk; bias = bk; out = k_up;  tr = false; }
    else                      { x = value; W = Wv; bias = bv; out = v_upT; tr = true;  }

    const int tid  = threadIdx.x;
    const int lane = tid & 31;
    const int w    = tid >> 5;
    const int wm   = w & 3;
    const int wn   = w >> 2;
    const int m0   = blockIdx.x * 64;
    const int n0   = blockIdx.y * 64;

    const int ldr = tid >> 2;          // 0..63
    const int ldc = (tid & 3) * 8;     // 0,8,16,24

    v8f acc0 = {}; v8f acc1 = {};

    for (int k0 = 0; k0 < DMODEL; k0 += 32) {
        __syncthreads();
        {
            const float4* xr = (const float4*)(x + (size_t)(m0 + ldr) * DMODEL + k0 + ldc);
            const float4* wr = (const float4*)(W + (size_t)(n0 + ldr) * DMODEL + k0 + ldc);
            float4 xa = xr[0], xb = xr[1];
            float4 wa = wr[0], wb = wr[1];
            unsigned int* xd = &XA[ldr * 16 + (ldc >> 1)];
            unsigned int* wd = &WB[ldr * 16 + (ldc >> 1)];
            xd[0] = pack_bf16(xa.x, xa.y); xd[1] = pack_bf16(xa.z, xa.w);
            xd[2] = pack_bf16(xb.x, xb.y); xd[3] = pack_bf16(xb.z, xb.w);
            wd[0] = pack_bf16(wa.x, wa.y); wd[1] = pack_bf16(wa.z, wa.w);
            wd[2] = pack_bf16(wb.x, wb.y); wd[3] = pack_bf16(wb.z, wb.w);
        }
        __syncthreads();

        v16bf a  = load_a_frag(XA, wm * 16, 16, 0, lane);
        v16bf b0 = load_b_frag(WB, wn * 32,      16, 0, lane);
        v16bf b1 = load_b_frag(WB, wn * 32 + 16, 16, 0, lane);
        acc0 = wmma_bf16(a, b0, acc0);
        acc1 = wmma_bf16(a, b1, acc1);
    }

    const int hi = lane >> 4, nl = lane & 15;
#pragma unroll
    for (int sub = 0; sub < 2; ++sub) {
        v8f c = sub ? acc1 : acc0;
#pragma unroll
        for (int e = 0; e < 8; ++e) {
            int n = n0 + wn * 32 + sub * 16 + nl;
            int m = m0 + wm * 16 + e + hi * 8;
            float v = c[e] + bias[n];
            int bidx = m >> 11, s = m & (SEQ - 1);
            int h = n >> 6,    d = n & 63;
            size_t idx = tr
                ? (((size_t)(bidx * HEADS + h) * DK + d) * SEQ + s)
                : (((size_t)(bidx * HEADS + h) * SEQ + s) * DK + d);
            out[idx] = f2bf_hw(v);
        }
    }
}

// ---------------------------------------------------------------------------
// Kernel 2: attention. grid (SEQ/64, H, B), block 256 (8 waves).
// K/V tiles streamed by the Tensor Data Mover into double-buffered LDS,
// pipelined one tile ahead of the WMMA/softmax compute.
// ---------------------------------------------------------------------------
__global__ __launch_bounds__(256) void attn_kernel(
    const unsigned short* __restrict__ q_up, const unsigned short* __restrict__ k_up,
    const unsigned short* __restrict__ v_upT, const unsigned char* __restrict__ mask,
    float* __restrict__ attn_out, float* __restrict__ o_ws)
{
    __shared__ __align__(16) unsigned int QT[64 * 32];      // [sq][d]  bf16
    __shared__ __align__(16) unsigned int KT[2][64 * 32];   // [sk][d]  bf16 (double buf)
    __shared__ __align__(16) unsigned int VT[2][64 * 32];   // [d][sk]  bf16 (double buf)
    __shared__ __align__(16) unsigned int PT[64 * 32];      // [sq][sk] bf16

    const int tid  = threadIdx.x;
    const int lane = tid & 31;
    const int w    = tid >> 5;
    const int wm   = w & 3;
    const int wn   = w >> 2;
    const int hi   = lane >> 4, nl = lane & 15;

    const int q0 = blockIdx.x * 64;
    const int h  = blockIdx.y;
    const int bb = blockIdx.z;
    const int bh = bb * HEADS + h;

    const unsigned short* kbase = k_up  + (size_t)bh * SEQ * DK;
    const unsigned short* vbase = v_upT + (size_t)bh * DK * SEQ;

    // ---- stage Q tile + prime pass-1 pipeline ----
#if defined(HAVE_TDM)
    if (w == 0) {
        tdm_load_2d((unsigned)(uintptr_t)(void*)QT,
                    q_up + (size_t)bh * SEQ * DK + (size_t)q0 * DK,
                    DK, SEQ, DK, 64, DK);
        tdm_load_2d((unsigned)(uintptr_t)(void*)KT[0], kbase, DK, SEQ, DK, 64, DK);
        __builtin_amdgcn_s_wait_tensorcnt(1);        // Q resident; K0 may stream
    }
#else
    {
        const uint4* qsrc = (const uint4*)(q_up) + ((size_t)bh * SEQ + q0) * (DK / 8);
#pragma unroll
        for (int i = 0; i < 2; ++i)
            ((uint4*)QT)[tid * 2 + i] = qsrc[tid * 2 + i];
    }
#endif
    __syncthreads();

    v16bf aQ0 = load_a_frag(QT, wm * 16, 32, 0,  lane);
    v16bf aQ1 = load_a_frag(QT, wm * 16, 32, 16, lane);

    float mrun[8], lrun[8];
#pragma unroll
    for (int e = 0; e < 8; ++e) { mrun[e] = -INFINITY; lrun[e] = 0.0f; }

    const size_t mbase = (size_t)bb * SEQ * SEQ;

    // ----------------- pass 1: softmax statistics -----------------
    for (int j = 0; j < NKT; ++j) {
        const int sk0 = j * 64;
        const int cur = j & 1;
#if defined(HAVE_TDM)
        if (w == 0) {
            if (j + 1 < NKT) {      // prefetch next K tile into alternate buffer
                tdm_load_2d((unsigned)(uintptr_t)(void*)KT[cur ^ 1],
                            kbase + (size_t)(sk0 + 64) * DK, DK, SEQ, DK, 64, DK);
                __builtin_amdgcn_s_wait_tensorcnt(1);   // tile j resident
            } else {
                __builtin_amdgcn_s_wait_tensorcnt(0);
            }
        }
#else
        {
            const uint4* ks = (const uint4*)(kbase) + (size_t)sk0 * (DK / 8);
#pragma unroll
            for (int i = 0; i < 2; ++i)
                ((uint4*)KT[cur])[tid * 2 + i] = ks[tid * 2 + i];
        }
#endif
        __syncthreads();                                 // (A) tile j ready

#pragma unroll
        for (int sub = 0; sub < 2; ++sub) {
            const int nb = wn * 32 + sub * 16;
            v8f c = {};
            c = wmma_bf16(aQ0, load_b_frag(KT[cur], nb, 32, 0,  lane), c);
            c = wmma_bf16(aQ1, load_b_frag(KT[cur], nb, 32, 16, lane), c);
            const int skg = sk0 + nb + nl;
            if (j + 1 < NKT)
                __builtin_prefetch(&mask[mbase + (size_t)(q0 + wm * 16 + hi * 8) * SEQ + skg + 64], 0, 1);
#pragma unroll
            for (int e = 0; e < 8; ++e) {
                const int sq = q0 + wm * 16 + e + hi * 8;
                float s = c[e] * 0.125f;
                if (mask[mbase + (size_t)sq * SEQ + skg]) s = -INFINITY;
                float tmax = grp16_max(s);
                float mnew = fmaxf(mrun[e], tmax);
                float p    = __expf(s - mnew);
                float ps   = grp16_sum(p);
                lrun[e] = lrun[e] * __expf(mrun[e] - mnew) + ps;
                mrun[e] = mnew;
            }
        }
        __syncthreads();                                 // (C) done reading KT[cur]
    }

    float inv[8];
#pragma unroll
    for (int e = 0; e < 8; ++e) inv[e] = 1.0f / lrun[e];

    v8f accO0 = {}; v8f accO1 = {};

    // ----------------- pass 2: write attn + PV -----------------
#if defined(HAVE_TDM)
    if (w == 0) {                                        // prime pipeline
        tdm_load_2d((unsigned)(uintptr_t)(void*)KT[0], kbase, DK, SEQ, DK, 64, DK);
        tdm_load_2d((unsigned)(uintptr_t)(void*)VT[0], vbase, SEQ, DK, 64, DK, SEQ);
    }
#endif
    for (int j = 0; j < NKT; ++j) {
        const int sk0 = j * 64;
        const int cur = j & 1;
#if defined(HAVE_TDM)
        if (w == 0) {
            if (j + 1 < NKT) {      // prefetch next K+V tiles into alternate buffers
                tdm_load_2d((unsigned)(uintptr_t)(void*)KT[cur ^ 1],
                            kbase + (size_t)(sk0 + 64) * DK, DK, SEQ, DK, 64, DK);
                tdm_load_2d((unsigned)(uintptr_t)(void*)VT[cur ^ 1],
                            vbase + sk0 + 64, SEQ, DK, 64, DK, SEQ);
                __builtin_amdgcn_s_wait_tensorcnt(2);    // tile j (K+V) resident
            } else {
                __builtin_amdgcn_s_wait_tensorcnt(0);
            }
        }
#else
        {
            const uint4* ks = (const uint4*)(kbase) + (size_t)sk0 * (DK / 8);
#pragma unroll
            for (int i = 0; i < 2; ++i)
                ((uint4*)KT[cur])[tid * 2 + i] = ks[tid * 2 + i];
            const int d = tid >> 2, part = tid & 3;
            const uint4* vs = (const uint4*)(vbase) +
                ((size_t)d * SEQ + sk0) / 8 + part * 2;
#pragma unroll
            for (int i = 0; i < 2; ++i)
                ((uint4*)VT[cur])[d * 8 + part * 2 + i] = vs[i];
        }
#endif
        __syncthreads();                                 // (A) tile j ready

#pragma unroll
        for (int sub = 0; sub < 2; ++sub) {
            const int nb = wn * 32 + sub * 16;
            v8f c = {};
            c = wmma_bf16(aQ0, load_b_frag(KT[cur], nb, 32, 0,  lane), c);
            c = wmma_bf16(aQ1, load_b_frag(KT[cur], nb, 32, 16, lane), c);
            const int skg = sk0 + nb + nl;
            if (j + 1 < NKT)
                __builtin_prefetch(&mask[mbase + (size_t)(q0 + wm * 16 + hi * 8) * SEQ + skg + 64], 0, 1);
#pragma unroll
            for (int e = 0; e < 8; ++e) {
                const int sq = q0 + wm * 16 + e + hi * 8;
                float s = c[e] * 0.125f;
                if (mask[mbase + (size_t)sq * SEQ + skg]) s = -INFINITY;
                float p = __expf(s - mrun[e]) * inv[e];
                attn_out[((size_t)bh * SEQ + sq) * SEQ + skg] = p;
                ((unsigned short*)PT)[(wm * 16 + e + hi * 8) * 64 + nb + nl] = f2bf_hw(p);
            }
        }
        __syncthreads();                                 // (B) PT ready

#pragma unroll
        for (int ks = 0; ks < 2; ++ks) {
            v16bf aP = load_a_frag(PT, wm * 16, 32, ks * 16, lane);
            v16bf b0 = load_b_frag(VT[cur], wn * 32,      32, ks * 16, lane);
            v16bf b1 = load_b_frag(VT[cur], wn * 32 + 16, 32, ks * 16, lane);
            accO0 = wmma_bf16(aP, b0, accO0);
            accO1 = wmma_bf16(aP, b1, accO1);
        }
        __syncthreads();                                 // (C) done with KT/VT[cur], PT
    }

    // attention output (pre-residual) as f32 [B,S,D]
#pragma unroll
    for (int sub = 0; sub < 2; ++sub) {
        v8f c = sub ? accO1 : accO0;
#pragma unroll
        for (int e = 0; e < 8; ++e) {
            const int d  = wn * 32 + sub * 16 + nl;
            const int sq = q0 + wm * 16 + e + hi * 8;
            o_ws[((size_t)bb * SEQ + sq) * DMODEL + h * DK + d] = c[e];
        }
    }
}

// ---------------------------------------------------------------------------
// Kernel 3: residual + OpenNMT LayerNorm (unbiased std). One block per row.
// ---------------------------------------------------------------------------
__global__ __launch_bounds__(256) void ln_kernel(
    const float* __restrict__ o_ws, const float* __restrict__ query,
    const float* __restrict__ a2, const float* __restrict__ b2,
    float* __restrict__ res_out)
{
    __shared__ float red[256];
    const int m   = blockIdx.x;
    const int tid = threadIdx.x;
    const size_t base = (size_t)m * DMODEL;

    float x0 = o_ws[base + tid]       + query[base + tid];
    float x1 = o_ws[base + 256 + tid] + query[base + 256 + tid];

    red[tid] = x0 + x1;
    __syncthreads();
    for (int s = 128; s > 0; s >>= 1) {
        if (tid < s) red[tid] += red[tid + s];
        __syncthreads();
    }
    float mu = red[0] * (1.0f / DMODEL);
    __syncthreads();

    float d0 = x0 - mu, d1 = x1 - mu;
    red[tid] = d0 * d0 + d1 * d1;
    __syncthreads();
    for (int s = 128; s > 0; s >>= 1) {
        if (tid < s) red[tid] += red[tid + s];
        __syncthreads();
    }
    float var  = red[0] * (1.0f / (DMODEL - 1));
    float rden = 1.0f / (sqrtf(var) + 1e-6f);

    res_out[base + tid]       = a2[tid]       * d0 * rden + b2[tid];
    res_out[base + 256 + tid] = a2[256 + tid] * d1 * rden + b2[256 + tid];
}

// ---------------------------------------------------------------------------
extern "C" void kernel_launch(void* const* d_in, const int* in_sizes, int n_in,
                              void* d_out, int out_size, void* d_ws, size_t ws_size,
                              hipStream_t stream) {
    const float* key   = (const float*)d_in[0];
    const float* value = (const float*)d_in[1];
    const float* query = (const float*)d_in[2];
    const unsigned char* mask = (const unsigned char*)d_in[3];
    const float* Wk = (const float*)d_in[4];
    const float* bk = (const float*)d_in[5];
    const float* Wv = (const float*)d_in[6];
    const float* bv = (const float*)d_in[7];
    const float* Wq = (const float*)d_in[8];
    const float* bq = (const float*)d_in[9];
    const float* a2 = (const float*)d_in[10];
    const float* b2 = (const float*)d_in[11];

    const size_t NPROJ = (size_t)BATCH * SEQ * DMODEL;   // 4,194,304
    unsigned short* q_up  = (unsigned short*)d_ws;
    unsigned short* k_up  = q_up + NPROJ;
    unsigned short* v_upT = k_up + NPROJ;
    float*          o_ws  = (float*)(v_upT + NPROJ);

    float* res_out  = (float*)d_out;
    float* attn_out = res_out + NPROJ;

    dim3 g1((BATCH * SEQ) / 64, DMODEL / 64, 3);
    proj_kernel<<<g1, 256, 0, stream>>>(key, value, query, Wk, bk, Wv, bv, Wq, bq,
                                        q_up, k_up, v_upT);

    dim3 g2(SEQ / 64, HEADS, BATCH);
    attn_kernel<<<g2, 256, 0, stream>>>(q_up, k_up, v_upT, mask, attn_out, o_ws);

    ln_kernel<<<BATCH * SEQ, 256, 0, stream>>>(o_ws, query, a2, b2, res_out);
}
